// MSHGAT_8435315769368
// MI455X (gfx1250) — compile-verified, matrix operand order checked by reading, not run on hardware
//
#include <hip/hip_runtime.h>
#include <hip/hip_bf16.h>

typedef __attribute__((ext_vector_type(16))) _Float16 v16h;
typedef __attribute__((ext_vector_type(8)))  float    v8f;

// Block tile: 64(M) x 128(N) x 32(K). 256 threads = 8 wave32s (2x4);
// each wave computes a 32x32 macro-tile = 2x2 WMMA 16x16 tiles.
#define BM 64
#define BN 128
#define BK 32

// ---------------------------------------------------------------------------
// Fragment loader: K-pairs are contiguous halfs in a 32-half row, so each
// VGPR of the fragment is one aligned dword; dwords form two contiguous
// 4-dword runs -> merges into ds_load_b128 pairs.
// ---------------------------------------------------------------------------
__device__ __forceinline__ v16h load_frag(const _Float16* __restrict__ row,
                                          int half_) {
    const unsigned* pu = (const unsigned*)row;
    union { unsigned u[8]; v16h v; } r;
    const int b = half_ * 4;
#pragma unroll
    for (int p = 0; p < 4; ++p) {
        r.u[p]     = pu[b + p];        // K pairs 0..7   (+8 for lane-half 1)
        r.u[p + 4] = pu[8 + b + p];    // K pairs 16..23 (+8 for lane-half 1)
    }
    return r.v;
}

// ---------------------------------------------------------------------------
// GEMM: C[M,N] = A[M,K] @ B[K,N] (+bias). f32 in/out, f16 WMMA inside.
// Requires: K % 32 == 0, N % 4 == 0, lda/ldb % 4 == 0 (callers guarantee).
// ---------------------------------------------------------------------------
__global__ __launch_bounds__(256)
void gemm_wmma_f16(const float* __restrict__ A, const float* __restrict__ Bm,
                   const float* __restrict__ bias, float* __restrict__ C,
                   int M, int N, int K, int lda, int ldb, int ldc)
{
    __shared__ _Float16 sA [BM][BK];   // row-major A tile   (4 KB)
    __shared__ _Float16 sBt[BN][BK];   // TRANSPOSED B tile  (8 KB)

    const int tid  = threadIdx.x;
    const int wave = tid >> 5;
    const int lane = tid & 31;
    const int wy = wave >> 2;              // 0..1 : M half
    const int wx = wave & 3;               // 0..3 : N quarter
    const int bm = blockIdx.y * BM;
    const int bn = blockIdx.x * BN;
    const int half_ = lane >> 4;
    const int mr    = lane & 15;

    // per-thread staging coordinates (fixed across K steps)
    const int ar  = tid >> 3;              // A row within tile (0..63), 2 rows, 32 apart
    const int ac4 = (tid & 7) << 2;        // A col (float4)
    const int br  = tid >> 5;              // B row within K tile (0..31), 8 apart? no: 4/thread below
    const int bc4 = (tid & 31) << 2;       // B col (float4)

    const bool full = (bm + BM <= M) && (bn + BN <= N);

    v8f acc[2][2] = {};

    for (int k0 = 0; k0 < K; k0 += BK) {
        // ---- load phase: issue all global_load_b128 before any use ----
        float4 va[2], vb[4];
        if (full) {
#pragma unroll
            for (int it = 0; it < 2; ++it)
                va[it] = *(const float4*)
                    &A[(size_t)(bm + ar + it * 32) * lda + k0 + ac4];
#pragma unroll
            for (int it = 0; it < 4; ++it)
                vb[it] = *(const float4*)
                    &Bm[(size_t)(k0 + br + it * 8) * ldb + bn + bc4];
            if (k0 + BK < K) {
                __builtin_prefetch(&A[(size_t)(bm + ar) * lda + k0 + BK], 0, 2);
                __builtin_prefetch(&Bm[(size_t)(k0 + BK + br) * ldb + bn + bc4], 0, 2);
            }
        } else {
#pragma unroll
            for (int it = 0; it < 2; ++it) {
                int gm = bm + ar + it * 32;
                va[it] = make_float4(0.f, 0.f, 0.f, 0.f);
                if (gm < M)
                    va[it] = *(const float4*)&A[(size_t)gm * lda + k0 + ac4];
            }
#pragma unroll
            for (int it = 0; it < 4; ++it) {
                int gn = bn + bc4;
                vb[it] = make_float4(0.f, 0.f, 0.f, 0.f);
                if (gn < N)
                    vb[it] = *(const float4*)
                        &Bm[(size_t)(k0 + br + it * 8) * ldb + gn];
            }
        }

        // ---- convert + store phase ----
#pragma unroll
        for (int it = 0; it < 2; ++it) {
            union { _Float16 h[4]; uint2 u; } pk;
            pk.h[0] = (_Float16)va[it].x; pk.h[1] = (_Float16)va[it].y;
            pk.h[2] = (_Float16)va[it].z; pk.h[3] = (_Float16)va[it].w;
            *(uint2*)&sA[ar + it * 32][ac4] = pk.u;
        }
#pragma unroll
        for (int it = 0; it < 4; ++it) {
            int r = br + it * 8;
            sBt[bc4 + 0][r] = (_Float16)vb[it].x;
            sBt[bc4 + 1][r] = (_Float16)vb[it].y;
            sBt[bc4 + 2][r] = (_Float16)vb[it].z;
            sBt[bc4 + 3][r] = (_Float16)vb[it].w;
        }
        __syncthreads();

        // ---- fragments + 2x2 WMMA ----
        v16h af[2], bf[2];
#pragma unroll
        for (int mi = 0; mi < 2; ++mi)
            af[mi] = load_frag(&sA[wy * 32 + mi * 16 + mr][0], half_);
#pragma unroll
        for (int ni = 0; ni < 2; ++ni)
            bf[ni] = load_frag(&sBt[wx * 32 + ni * 16 + mr][0], half_);
#pragma unroll
        for (int mi = 0; mi < 2; ++mi)
#pragma unroll
            for (int ni = 0; ni < 2; ++ni)
                acc[mi][ni] = __builtin_amdgcn_wmma_f32_16x16x32_f16(
                    false, af[mi], false, bf[ni], (short)0,
                    acc[mi][ni], false, false);
        __syncthreads();
    }

    // ---- store: VGPR r -> row (r + 8*half), col = lane&15 ----
#pragma unroll
    for (int ni = 0; ni < 2; ++ni) {
        int ncol = bn + wx * 32 + ni * 16 + (lane & 15);
        if (ncol >= N) continue;
        float badd = bias ? bias[ncol] : 0.0f;
#pragma unroll
        for (int mi = 0; mi < 2; ++mi) {
#pragma unroll
            for (int r = 0; r < 8; ++r) {
                int mrow = bm + wy * 32 + mi * 16 + r + half_ * 8;
                if (mrow < M)
                    C[(size_t)mrow * ldc + ncol] = acc[mi][ni][r] + badd;
            }
        }
    }
}

// ---------------------------------------------------------------------------
// Degrees with self loops, then in-place dinv = rsqrt(max(deg,1e-12))
// ---------------------------------------------------------------------------
__global__ void deg_init(float* deg, int Nn) {
    int i = blockIdx.x * blockDim.x + threadIdx.x;
    if (i < Nn) deg[i] = 1.0f;                 // self loop
}
__global__ void deg_edges(const int* __restrict__ dst, float* deg, int E) {
    int e = blockIdx.x * blockDim.x + threadIdx.x;
    if (e < E) atomicAdd(&deg[dst[e]], 1.0f);
}
__global__ void deg_rsqrt(float* deg, int Nn) {
    int i = blockIdx.x * blockDim.x + threadIdx.x;
    if (i < Nn) deg[i] = rsqrtf(fmaxf(deg[i], 1e-12f));
}

// out[i,:] = bias[:]  (row broadcast init before scatter accumulation)
__global__ void row_init(float* out, const float* __restrict__ bias,
                         int Nn, int ld) {
    size_t total = (size_t)Nn * ld;
    for (size_t i = (size_t)blockIdx.x * blockDim.x + threadIdx.x;
         i < total; i += (size_t)gridDim.x * blockDim.x)
        out[i] = bias[i % ld];
}

// one wave per edge; self loops appended as edges [E, E+Nn). ld % 4 == 0.
__global__ __launch_bounds__(256)
void edge_scatter(const int* __restrict__ src, const int* __restrict__ dst,
                  int E, int Nn, const float* __restrict__ dinv,
                  const float* __restrict__ h, float* out, int ld)
{
    const int wpb   = blockDim.x >> 5;
    const int lane  = threadIdx.x & 31;
    const int wave0 = blockIdx.x * wpb + (threadIdx.x >> 5);
    const int nw    = gridDim.x * wpb;
    const int total = E + Nn;
    for (int e = wave0; e < total; e += nw) {
        int s, d;
        if (e < E) { s = src[e]; d = dst[e]; }
        else       { s = e - E;  d = s; }
        float c = dinv[s] * dinv[d];
        const float* hs = h   + (size_t)s * ld;
        float*       od = out + (size_t)d * ld;
        for (int j = lane * 4; j < ld; j += 128) {
            float4 hv = *(const float4*)&hs[j];
            atomicAdd(&od[j + 0], hv.x * c);
            atomicAdd(&od[j + 1], hv.y * c);
            atomicAdd(&od[j + 2], hv.z * c);
            atomicAdd(&od[j + 3], hv.w * c);
        }
    }
}

// ---------------------------------------------------------------------------
// BatchNorm (training stats over N nodes, per column of 64)
// ---------------------------------------------------------------------------
__global__ __launch_bounds__(256)
void bn_stats(const float* __restrict__ h, int Nn, float* mu, float* var) {
    const int d = blockIdx.x;                  // 0..63
    __shared__ float ss[256], sq[256];
    float s = 0.0f, q = 0.0f;
    for (int i = threadIdx.x; i < Nn; i += blockDim.x) {
        float x = h[(size_t)i * 64 + d];
        s += x; q += x * x;
    }
    ss[threadIdx.x] = s; sq[threadIdx.x] = q;
    __syncthreads();
    for (int off = 128; off > 0; off >>= 1) {
        if (threadIdx.x < off) {
            ss[threadIdx.x] += ss[threadIdx.x + off];
            sq[threadIdx.x] += sq[threadIdx.x + off];
        }
        __syncthreads();
    }
    if (threadIdx.x == 0) {
        float m = ss[0] / (float)Nn;
        mu[d]  = m;
        var[d] = sq[0] / (float)Nn - m * m;    // population var (ddof=0)
    }
}

__global__ void bn_norm(float* h, int Nn, const float* __restrict__ mu,
                        const float* __restrict__ var,
                        const float* __restrict__ gamma,
                        const float* __restrict__ beta) {
    size_t total = (size_t)Nn * 64;
    for (size_t i = (size_t)blockIdx.x * blockDim.x + threadIdx.x;
         i < total; i += (size_t)gridDim.x * blockDim.x) {
        int d = (int)(i & 63);
        h[i] = (h[i] - mu[d]) * rsqrtf(var[d] + 1e-5f) * gamma[d] + beta[d];
    }
}

// W_out [72,U] -> padded [96,U] with zero rows (removes all K guards in GEMM)
__global__ void pad_wout(const float* __restrict__ Wout, float* __restrict__ pW,
                         int Kin, int Kpad, int U) {
    size_t total = (size_t)Kpad * U;
    for (size_t i = (size_t)blockIdx.x * blockDim.x + threadIdx.x;
         i < total; i += (size_t)gridDim.x * blockDim.x) {
        int k = (int)(i / U);
        pW[i] = (k < Kin) ? Wout[i] : 0.0f;
    }
}

// ---------------------------------------------------------------------------
// Fusion gate + positional concat + PAD mask. One block per (b,l).
// Writes padded rows of stride 96 (cols 72..95 zeroed) for the clean GEMM.
// ---------------------------------------------------------------------------
__global__ __launch_bounds__(128)
void fuse_kernel(const int* __restrict__ seq, const float* __restrict__ hidden,
                 const float* __restrict__ dyn,
                 const float* __restrict__ fW1, const float* __restrict__ fb1,
                 const float* __restrict__ fW2, const float* __restrict__ fb2,
                 const float* __restrict__ pos, float* __restrict__ finalbuf,
                 int Lq)
{
    const int p = blockIdx.x;
    const int l = p % Lq;
    const int u = seq[p];
    __shared__ float xs[2][64];
    __shared__ float red[128];
    __shared__ float score[2];

    const int tid = threadIdx.x;
    const int s   = tid >> 6;                  // source 0/1
    const int d   = tid & 63;
    xs[s][d] = (s == 0) ? hidden[(size_t)u * 64 + d]
                        : dyn[(size_t)u * 64 + d];
    __syncthreads();

    float a = fb1[d];
    for (int k = 0; k < 64; ++k) a += xs[s][k] * fW1[k * 64 + d];
    red[tid] = tanhf(a) * fW2[d];
    __syncthreads();
    if (d == 0) {
        float acc = 0.0f;
        for (int k = 0; k < 64; ++k) acc += red[s * 64 + k];
        score[s] = acc + fb2[0];
    }
    __syncthreads();

    float m  = fmaxf(score[0], score[1]);
    float e0 = __expf(score[0] - m), e1 = __expf(score[1] - m);
    float g0 = e0 / (e0 + e1),       g1 = e1 / (e0 + e1);

    float keep = (u != 0) ? 1.0f : 0.0f;
    float* row = finalbuf + (size_t)p * 96;
    if (s == 0)
        row[d] = keep * (g0 * xs[0][d] + g1 * xs[1][d]);
    if (tid >= 64 && tid < 72)
        row[tid] = keep * pos[(size_t)l * 8 + (tid - 64)];
    else if (tid >= 72 && tid < 96)
        row[tid] = 0.0f;                       // K padding
}

// ---------------------------------------------------------------------------
// Previous-user mask: .set(-1000) semantics via per-(b,t) LDS bitmask.
// idx = tri ? seq[b,t'] : 0, so t' > t contributes user 0.
// ---------------------------------------------------------------------------
#define MASK_WORDS 1568   // covers up to 50176 users
__global__ __launch_bounds__(256)
void mask_kernel(const int* __restrict__ seq, float* __restrict__ logits,
                 int Lq, int U)
{
    const int p = blockIdx.x;                  // (b,t) flattened
    const int b = p / Lq, t = p % Lq;
    __shared__ unsigned bits[MASK_WORDS];
    const int words = (U + 31) >> 5;
    for (int i = threadIdx.x; i < words; i += blockDim.x) bits[i] = 0u;
    __syncthreads();
    for (int tp = threadIdx.x; tp < Lq; tp += blockDim.x) {
        int u = (tp <= t) ? seq[b * Lq + tp] : 0;
        atomicOr(&bits[u >> 5], 1u << (u & 31));
    }
    __syncthreads();
    float* row = logits + (size_t)p * U;
    for (int u = threadIdx.x; u < U; u += blockDim.x)
        if ((bits[u >> 5] >> (u & 31)) & 1u) row[u] -= 1000.0f;
}

// ---------------------------------------------------------------------------
extern "C" void kernel_launch(void* const* d_in, const int* in_sizes, int n_in,
                              void* d_out, int out_size, void* d_ws, size_t ws_size,
                              hipStream_t stream)
{
    const int*   seq   = (const int*)  d_in[0];
    const int*   edge  = (const int*)  d_in[1];
    const float* emb   = (const float*)d_in[2];
    const float* dyn   = (const float*)d_in[3];
    const float* W1    = (const float*)d_in[4];
    const float* b1    = (const float*)d_in[5];
    const float* W2    = (const float*)d_in[6];
    const float* b2    = (const float*)d_in[7];
    const float* gamma = (const float*)d_in[8];
    const float* beta  = (const float*)d_in[9];
    const float* fW1   = (const float*)d_in[10];
    const float* fb1   = (const float*)d_in[11];
    const float* fW2   = (const float*)d_in[12];
    const float* fb2   = (const float*)d_in[13];
    const float* pos   = (const float*)d_in[14];
    const float* Wout  = (const float*)d_in[15];
    const float* bout  = (const float*)d_in[16];

    const int D  = 64, D2 = 128, Lq = 200, KP = 96;
    const int Nn = in_sizes[2] / D;            // 50000
    const int E  = in_sizes[1] / 2;            // 800000
    const int BL = in_sizes[0];                // 1600
    const int U  = in_sizes[16];               // 50000
    const int* src = edge;
    const int* dst = edge + E;

    // workspace layout (floats)
    float* ws       = (float*)d_ws;
    float* deg      = ws;                          // Nn
    float* bufA     = deg + Nn;                    // Nn*128
    float* bufB     = bufA + (size_t)Nn * D2;      // Nn*128
    float* mu       = bufB + (size_t)Nn * D2;      // 64
    float* var      = mu + 64;                     // 64
    float* finalbuf = var + 64;                    // BL*96
    float* padW     = finalbuf + (size_t)BL * KP;  // 96*U
    float* logits   = (float*)d_out;               // BL*U

    // --- symmetric-norm degrees ---
    deg_init  <<<(Nn + 255) / 256, 256, 0, stream>>>(deg, Nn);
    deg_edges <<<(E  + 255) / 256, 256, 0, stream>>>(dst, deg, E);
    deg_rsqrt <<<(Nn + 255) / 256, 256, 0, stream>>>(deg, Nn);

    // --- pad W_out rows 72..95 with zeros (enables guard-free K loop) ---
    pad_wout<<<2048, 256, 0, stream>>>(Wout, padW, D + 8, KP, U);

    // --- GCN layer 1: h1 = emb @ W1 ; agg1 = scatter(h1)+b1 ---
    {
        dim3 g((D2 + BN - 1) / BN, (Nn + BM - 1) / BM);
        gemm_wmma_f16<<<g, 256, 0, stream>>>(emb, W1, nullptr, bufA,
                                             Nn, D2, D, D, D2, D2);
    }
    row_init<<<2048, 256, 0, stream>>>(bufB, b1, Nn, D2);
    edge_scatter<<<8192, 256, 0, stream>>>(src, dst, E, Nn, deg, bufA, bufB, D2);

    // --- GCN layer 2: h2 = agg1 @ W2 ; agg2 = scatter(h2)+b2 ---
    {
        dim3 g((D + BN - 1) / BN, (Nn + BM - 1) / BM);
        gemm_wmma_f16<<<g, 256, 0, stream>>>(bufB, W2, nullptr, bufA,
                                             Nn, D, D2, D2, D, D);
    }
    row_init<<<2048, 256, 0, stream>>>(bufB, b2, Nn, D);
    edge_scatter<<<8192, 256, 0, stream>>>(src, dst, E, Nn, deg, bufA, bufB, D);

    // --- BatchNorm (training-mode stats) in place on bufB ---
    bn_stats<<<64, 256, 0, stream>>>(bufB, Nn, mu, var);
    bn_norm <<<2048, 256, 0, stream>>>(bufB, Nn, mu, var, gamma, beta);

    // --- fusion gate + pos concat + PAD mask -> finalbuf [BL,96] ---
    fuse_kernel<<<BL, 128, 0, stream>>>(seq, bufB, dyn, fW1, fb1, fW2, fb2,
                                        pos, finalbuf, Lq);

    // --- logits = finalbuf @ padW + b_out (HBM-store-bound stage) ---
    {
        dim3 g((U + BN - 1) / BN, (BL + BM - 1) / BM);
        gemm_wmma_f16<<<g, 256, 0, stream>>>(finalbuf, padW, bout, logits,
                                             BL, U, KP, KP, U, U);
    }

    // --- previous-user mask ---
    mask_kernel<<<BL, 256, 0, stream>>>(seq, logits, Lq, U);

    (void)n_in; (void)out_size; (void)ws_size;
}
